// DecoderLSTM_79328045957728
// MI455X (gfx1250) — matrix operand order repaired and checked by reading, not compiled, not run on hardware
//
#include <hip/hip_runtime.h>
#include <hip/hip_bf16.h>

// ---------------------------------------------------------------------------
// DecoderLSTM with Bahdanau attention for MI455X (gfx1250, wave32, WMMA).
// Recurrent GEMMs use v_wmma_f32_16x16x32_f16 with f32 accumulation.
// One wave owns a full 32-row (both M-tiles) x 16-col C strip so the dominant
// B weight stream (fc_W: 32.8 MB f16, L2-resident) is read exactly once.
// Inner loop is software-pipelined: next K-tile's loads issue before the
// current tile's WMMAs consume their registers.
// ---------------------------------------------------------------------------

typedef __attribute__((ext_vector_type(16))) _Float16 v16h;
typedef __attribute__((ext_vector_type(8)))  float    v8f;

// Problem constants (match reference).
#define BB   32
#define SS   400
#define TT   100
#define TM1  99
#define EE   256
#define HH   512
#define AA   512
#define VV   32000
#define GG   2048          // 4*H
#define KGATE 1280         // E + H (inp) + H (hidden)

#define NKT_FC    16       // K=512 -> 16 k-tiles of 32
#define NNT_FC    2000     // N=32000 -> 2000 n-tiles of 16
#define NKT_G     40       // K=1280 -> 40 k-tiles
#define NNT_G     128      // N=2048 -> 128 n-tiles
#define NMT       2        // M=32 -> 2 m-tiles

__device__ __forceinline__ unsigned int pack_f16x2(float lo, float hi) {
    _Float16 l = (_Float16)lo;
    _Float16 h = (_Float16)hi;
    unsigned short lu, hu;
    __builtin_memcpy(&lu, &l, 2);
    __builtin_memcpy(&hu, &h, 2);
    return (unsigned int)lu | ((unsigned int)hu << 16);
}

__device__ __forceinline__ float sigmoidf_(float x) {
    return 1.0f / (1.0f + __expf(-x));
}

// ---------------------------------------------------------------------------
// Pack A matrix [M,K] f32 row-major into WMMA 16-bit A tile layout.
// One thread per (mtile, ktile, lane) slot; writes 8 dwords (16 f16).
// CDNA5 ISA: lanes 0-15: M=lane, K = {0..7, 16..23}; lanes 16-31:
// M=lane-16, K = {8..15, 24..31}; 2 K-values per dword.
// ---------------------------------------------------------------------------
__global__ void pack_a_f16(const float* __restrict__ src, int ldk,
                           unsigned int* __restrict__ dst,
                           int n_mt, int n_kt) {
    int slot = blockIdx.x * blockDim.x + threadIdx.x;
    if (slot >= n_mt * n_kt * 32) return;
    int lane = slot & 31;
    int tile = slot >> 5;
    int kt = tile % n_kt;
    int mt = tile / n_kt;
    int m  = mt * 16 + (lane & 15);
    int kb = kt * 32 + ((lane & 16) ? 8 : 0);
    unsigned int* out = dst + (size_t)slot * 8;
#pragma unroll
    for (int d = 0; d < 8; ++d) {
        int k = kb + (d < 4 ? 2 * d : 16 + 2 * (d - 4));
        out[d] = pack_f16x2(src[(size_t)m * ldk + k],
                            src[(size_t)m * ldk + k + 1]);
    }
}

// ---------------------------------------------------------------------------
// Pack B matrix from W-style [N,K] f32 (C = A * W^T) into WMMA B layout.
// B[k][n] = W[n][k]. Lane = column n; lanes 0-15 K={0..7,16..23}, lanes 16-31 K={8..15,24..31}.
// ---------------------------------------------------------------------------
__global__ void pack_b_nk_f16(const float* __restrict__ src, int ldk,
                              unsigned int* __restrict__ dst,
                              int n_nt, int n_kt) {
    int slot = blockIdx.x * blockDim.x + threadIdx.x;
    if (slot >= n_nt * n_kt * 32) return;
    int lane = slot & 31;
    int tile = slot >> 5;
    int kt = tile % n_kt;
    int nt = tile / n_kt;
    int n  = nt * 16 + (lane & 15);
    int kb = kt * 32 + ((lane & 16) ? 8 : 0);
    unsigned int* out = dst + (size_t)slot * 8;
#pragma unroll
    for (int d = 0; d < 8; ++d) {
        int k = kb + (d < 4 ? 2 * d : 16 + 2 * (d - 4));
        out[d] = pack_f16x2(src[(size_t)n * ldk + k],
                            src[(size_t)n * ldk + k + 1]);
    }
}

// Same, but B is concat(W_ih [2048,768], W_hh [2048,512]) along K (K=1280).
__global__ void pack_b_gates_f16(const float* __restrict__ wih,
                                 const float* __restrict__ whh,
                                 unsigned int* __restrict__ dst) {
    int slot = blockIdx.x * blockDim.x + threadIdx.x;
    if (slot >= NNT_G * NKT_G * 32) return;
    int lane = slot & 31;
    int tile = slot >> 5;
    int kt = tile % NKT_G;
    int nt = tile / NKT_G;
    int n  = nt * 16 + (lane & 15);
    int kb = kt * 32 + ((lane & 16) ? 8 : 0);
    unsigned int* out = dst + (size_t)slot * 8;
#pragma unroll
    for (int d = 0; d < 8; ++d) {
        int k  = kb + (d < 4 ? 2 * d : 16 + 2 * (d - 4));
        float lo = (k     < 768) ? wih[(size_t)n * 768 + k]
                                 : whh[(size_t)n * 512 + (k - 768)];
        float hi = (k + 1 < 768) ? wih[(size_t)n * 768 + k + 1]
                                 : whh[(size_t)n * 512 + (k + 1 - 768)];
        out[d] = pack_f16x2(lo, hi);
    }
}

// A-pack for the gate GEMM: rows are batch, K = [emb(256) | context(512) | h(512)].
__device__ __forceinline__ float gate_input(const float* __restrict__ emb,
                                            const float* __restrict__ ctx,
                                            const float* __restrict__ h,
                                            int m, int t, int k) {
    if (k < EE)       return emb[((size_t)m * TM1 + t) * EE + k];
    if (k < EE + HH)  return ctx[(size_t)m * HH + (k - EE)];
    return h[(size_t)m * HH + (k - EE - HH)];
}

__global__ void pack_a_gates_f16(const float* __restrict__ emb,
                                 const float* __restrict__ ctx,
                                 const float* __restrict__ h,
                                 unsigned int* __restrict__ dst, int t) {
    int slot = blockIdx.x * blockDim.x + threadIdx.x;
    if (slot >= NMT * NKT_G * 32) return;
    int lane = slot & 31;
    int tile = slot >> 5;
    int kt = tile % NKT_G;
    int mt = tile / NKT_G;
    int m  = mt * 16 + (lane & 15);
    int kb = kt * 32 + ((lane & 16) ? 8 : 0);
    unsigned int* out = dst + (size_t)slot * 8;
#pragma unroll
    for (int d = 0; d < 8; ++d) {
        int k = kb + (d < 4 ? 2 * d : 16 + 2 * (d - 4));
        out[d] = pack_f16x2(gate_input(emb, ctx, h, m, t, k),
                            gate_input(emb, ctx, h, m, t, k + 1));
    }
}

// ---------------------------------------------------------------------------
// WMMA GEMM, one wave per 16-wide N strip covering all M=32 rows:
//   C[0:32, nt*16:(nt+1)*16] = A[0:32, :] * B[:, nt*16:(nt+1)*16] + bias
// Two accumulators (M-tiles 0/1) share each loaded B tile -> B read once.
// Software-pipelined: loads for k-tile kt+1 are issued before the WMMAs of
// k-tile kt, so s_wait_loadcnt can overlap L2 latency with matrix math.
// A/B pre-packed as [tile][lane][8 dwords] -> one 32B load per lane per tile.
// ---------------------------------------------------------------------------
__global__ __launch_bounds__(32)
void wmma_gemm_f16(const unsigned int* __restrict__ Apack,
                   const unsigned int* __restrict__ Bpack,
                   const float* __restrict__ bias,
                   float* __restrict__ C, long long ldc,
                   int n_kt) {
    const int nt = blockIdx.x;
    const int lane = threadIdx.x;

    v8f acc0, acc1;
    float bv = bias ? bias[nt * 16 + (lane & 15)] : 0.0f;
#pragma unroll
    for (int r = 0; r < 8; ++r) { acc0[r] = bv; acc1[r] = bv; }

    const unsigned int* aptr0 = Apack + (size_t)lane * 8;                    // M-tile 0
    const unsigned int* aptr1 = Apack + (size_t)n_kt * 256 + (size_t)lane * 8; // M-tile 1
    const unsigned int* bptr  = Bpack + ((size_t)nt * n_kt) * 256 + (size_t)lane * 8;

    // Prologue: load k-tile 0.
    v16h b_cur  = *reinterpret_cast<const v16h*>(bptr);
    v16h a0_cur = *reinterpret_cast<const v16h*>(aptr0);
    v16h a1_cur = *reinterpret_cast<const v16h*>(aptr1);

    for (int kt = 0; kt < n_kt - 1; ++kt) {
        bptr  += 256;
        aptr0 += 256;
        aptr1 += 256;
        if (kt + 2 < n_kt) {
            __builtin_prefetch(bptr + 256, 0, 3);   // global_prefetch_b8, kt+2
        }
        // Issue next tile's loads before consuming the current registers.
        v16h b_nxt  = *reinterpret_cast<const v16h*>(bptr);
        v16h a0_nxt = *reinterpret_cast<const v16h*>(aptr0);
        v16h a1_nxt = *reinterpret_cast<const v16h*>(aptr1);

        acc0 = __builtin_amdgcn_wmma_f32_16x16x32_f16(
            false, a0_cur, false, b_cur, (short)0, acc0, false, false);
        acc1 = __builtin_amdgcn_wmma_f32_16x16x32_f16(
            false, a1_cur, false, b_cur, (short)0, acc1, false, false);

        b_cur = b_nxt; a0_cur = a0_nxt; a1_cur = a1_nxt;
    }
    acc0 = __builtin_amdgcn_wmma_f32_16x16x32_f16(
        false, a0_cur, false, b_cur, (short)0, acc0, false, false);
    acc1 = __builtin_amdgcn_wmma_f32_16x16x32_f16(
        false, a1_cur, false, b_cur, (short)0, acc1, false, false);

    // D layout: VGPR r holds M = r (+8 for lanes 16-31), N = nt*16 + lane%16.
    long long n = (long long)nt * 16 + (lane & 15);
    int mbase = (lane & 16) ? 8 : 0;
#pragma unroll
    for (int r = 0; r < 8; ++r) {
        C[(long long)(mbase + r) * ldc + n]      = acc0[r];
        C[(long long)(16 + mbase + r) * ldc + n] = acc1[r];
    }
}

// ---------------------------------------------------------------------------
// One-time: enc_proj[b,s,:] = encoder_outputs[b,s,:] @ att_Wenc + att_benc.
// Plain f32 (off critical path; keeps softmax inputs full precision).
// ---------------------------------------------------------------------------
__global__ __launch_bounds__(256)
void enc_proj_kernel(const float* __restrict__ enc,
                     const float* __restrict__ Wenc,   // [H, A]
                     const float* __restrict__ benc,
                     float* __restrict__ out) {
    const int m = blockIdx.x;                // 0 .. B*S-1
    __shared__ float row[HH];
    for (int k = threadIdx.x; k < HH; k += 256)
        row[k] = enc[(size_t)m * HH + k];
    __syncthreads();
    for (int n = threadIdx.x; n < AA; n += 256) {
        float acc = benc[n];
        for (int k = 0; k < HH; ++k)
            acc += row[k] * Wenc[(size_t)k * AA + n];
        out[(size_t)m * AA + n] = acc;
    }
}

// One-time: emb[b,t,:] = embedding[summary'[b,t]] with summary_len-1 zeroing.
__global__ void embed_gather_kernel(const int* __restrict__ summary,
                                    const int* __restrict__ summary_len,
                                    const float* __restrict__ embedding,
                                    float* __restrict__ emb) {
    int idx = blockIdx.x * blockDim.x + threadIdx.x;
    if (idx >= BB * TM1 * EE) return;
    int e = idx % EE;
    int t = (idx / EE) % TM1;
    int b = idx / (EE * TM1);
    int tok = summary[b * TT + t];
    if (t == summary_len[b] - 1) tok = 0;
    emb[idx] = embedding[(size_t)tok * EE + e];
}

// Combined LSTM gate bias (b_ih + b_hh).
__global__ void bias_combine_kernel(const float* __restrict__ b0,
                                    const float* __restrict__ b1,
                                    float* __restrict__ out) {
    int i = blockIdx.x * blockDim.x + threadIdx.x;
    if (i < GG) out[i] = b0[i] + b1[i];
}

// ---------------------------------------------------------------------------
// Per-step Bahdanau attention: one block per batch row, 256 threads (8 waves).
// ---------------------------------------------------------------------------
__global__ __launch_bounds__(256)
void attention_kernel(const float* __restrict__ h,
                      const float* __restrict__ enc,       // [B,S,H]
                      const float* __restrict__ enc_proj,  // [B,S,A]
                      const float* __restrict__ Wdec,      // [H,A]
                      const float* __restrict__ bdec,
                      const float* __restrict__ att_v,
                      float* __restrict__ context) {
    const int b = blockIdx.x;
    __shared__ float hs[HH];
    __shared__ float dp[AA];
    __shared__ float sc[SS];
    __shared__ float redmax[8];
    __shared__ float redsum[8];
    const int tid  = threadIdx.x;
    const int lane = tid & 31;
    const int wave = tid >> 5;

    for (int k = tid; k < HH; k += 256) hs[k] = h[b * HH + k];
    __syncthreads();

    // dec_proj = h @ Wdec + bdec
    for (int a = tid; a < AA; a += 256) {
        float acc = bdec[a];
        for (int k = 0; k < HH; ++k) acc += hs[k] * Wdec[(size_t)k * AA + a];
        dp[a] = acc;
    }
    __syncthreads();

    // scores[s] = sum_a tanh(enc_proj + dec_proj) * v   (one wave per s)
    const float* ep = enc_proj + (size_t)b * SS * AA;
    for (int s = wave; s < SS; s += 8) {
        float p = 0.0f;
        const float* eps = ep + (size_t)s * AA;
        for (int a = lane; a < AA; a += 32)
            p += tanhf(eps[a] + dp[a]) * att_v[a];
        for (int off = 16; off; off >>= 1) p += __shfl_xor(p, off, 32);
        if (lane == 0) sc[s] = p;
    }
    __syncthreads();

    // softmax over S=400
    float mx = -1e30f;
    for (int s = tid; s < SS; s += 256) mx = fmaxf(mx, sc[s]);
    for (int off = 16; off; off >>= 1) mx = fmaxf(mx, __shfl_xor(mx, off, 32));
    if (lane == 0) redmax[wave] = mx;
    __syncthreads();
    mx = redmax[0];
    for (int w = 1; w < 8; ++w) mx = fmaxf(mx, redmax[w]);

    float ls = 0.0f;
    for (int s = tid; s < SS; s += 256) {
        float e = __expf(sc[s] - mx);
        sc[s] = e;
        ls += e;
    }
    for (int off = 16; off; off >>= 1) ls += __shfl_xor(ls, off, 32);
    if (lane == 0) redsum[wave] = ls;
    __syncthreads();
    float tot = 0.0f;
    for (int w = 0; w < 8; ++w) tot += redsum[w];
    const float inv = 1.0f / tot;

    // context = alpha @ encoder_outputs
    for (int j = tid; j < HH; j += 256) {
        float acc = 0.0f;
        const float* eb = enc + (size_t)b * SS * HH + j;
        for (int s = 0; s < SS; ++s) acc += sc[s] * eb[(size_t)s * HH];
        context[b * HH + j] = acc * inv;
    }
}

// LSTM pointwise update (gates order i,f,g,o per jnp.split).
__global__ void lstm_pointwise_kernel(const float* __restrict__ gates, // [B,4H]
                                      float* __restrict__ h,
                                      float* __restrict__ c) {
    int idx = blockIdx.x * blockDim.x + threadIdx.x;
    if (idx >= BB * HH) return;
    int m = idx >> 9;          // /512
    int j = idx & (HH - 1);
    const float* g = gates + (size_t)m * GG;
    float gi = g[j], gf = g[HH + j], gg = g[2 * HH + j], go = g[3 * HH + j];
    float cn = sigmoidf_(gf) * c[idx] + sigmoidf_(gi) * tanhf(gg);
    float hn = sigmoidf_(go) * tanhf(cn);
    c[idx] = cn;
    h[idx] = hn;
}

// ---------------------------------------------------------------------------
// Host-side orchestration.
// ---------------------------------------------------------------------------
extern "C" void kernel_launch(void* const* d_in, const int* in_sizes, int n_in,
                              void* d_out, int out_size, void* d_ws, size_t ws_size,
                              hipStream_t stream) {
    const float* enc       = (const float*)d_in[0];
    const float* hidden0   = (const float*)d_in[1];
    const float* cell0     = (const float*)d_in[2];
    const int*   summary   = (const int*)  d_in[3];
    const int*   summarylen= (const int*)  d_in[4];
    const float* embedding = (const float*)d_in[5];
    const float* W_ih      = (const float*)d_in[6];
    const float* b_ih      = (const float*)d_in[7];
    const float* W_hh      = (const float*)d_in[8];
    const float* b_hh      = (const float*)d_in[9];
    const float* att_Wenc  = (const float*)d_in[10];
    const float* att_benc  = (const float*)d_in[11];
    const float* att_Wdec  = (const float*)d_in[12];
    const float* att_bdec  = (const float*)d_in[13];
    const float* att_v     = (const float*)d_in[14];
    const float* fc_W      = (const float*)d_in[15];
    const float* fc_b      = (const float*)d_in[16];
    float* out = (float*)d_out;

    // Workspace carve-up (bytes, all 256B-aligned sizes).
    char* base = (char*)d_ws;
    size_t off = 0;
    auto carve = [&](size_t bytes) { char* p = base + off; off += bytes; return p; };
    unsigned int* Bfc   = (unsigned int*)carve((size_t)NNT_FC * NKT_FC * 256 * 4); // 32.8 MB
    unsigned int* Bg    = (unsigned int*)carve((size_t)NNT_G  * NKT_G  * 256 * 4); //  5.2 MB
    float* enc_proj     = (float*)carve((size_t)BB * SS * AA * 4);                 // 26.2 MB
    float* emb          = (float*)carve((size_t)BB * TM1 * EE * 4);                //  3.2 MB
    float* h            = (float*)carve((size_t)BB * HH * 4);
    float* c            = (float*)carve((size_t)BB * HH * 4);
    float* ctx          = (float*)carve((size_t)BB * HH * 4);
    float* gates        = (float*)carve((size_t)BB * GG * 4);
    unsigned int* Ag    = (unsigned int*)carve((size_t)NMT * NKT_G * 256 * 4);
    unsigned int* Afc   = (unsigned int*)carve((size_t)NMT * NKT_FC * 256 * 4);
    float* gbias        = (float*)carve((size_t)GG * 4);
    (void)ws_size; (void)n_in; (void)in_sizes; (void)out_size;

    // ---- one-time setup (all on stream; graph-capture safe) ----
    hipMemcpyAsync(h, hidden0, (size_t)BB * HH * 4, hipMemcpyDeviceToDevice, stream);
    hipMemcpyAsync(c, cell0,   (size_t)BB * HH * 4, hipMemcpyDeviceToDevice, stream);

    bias_combine_kernel<<<(GG + 255) / 256, 256, 0, stream>>>(b_ih, b_hh, gbias);

    {   // fc_W [V,H] -> B tiles
        int slots = NNT_FC * NKT_FC * 32;
        pack_b_nk_f16<<<(slots + 255) / 256, 256, 0, stream>>>(fc_W, HH, Bfc, NNT_FC, NKT_FC);
    }
    {   // [W_ih | W_hh] -> B tiles
        int slots = NNT_G * NKT_G * 32;
        pack_b_gates_f16<<<(slots + 255) / 256, 256, 0, stream>>>(W_ih, W_hh, Bg);
    }

    enc_proj_kernel<<<BB * SS, 256, 0, stream>>>(enc, att_Wenc, att_benc, enc_proj);

    {
        int n = BB * TM1 * EE;
        embed_gather_kernel<<<(n + 255) / 256, 256, 0, stream>>>(summary, summarylen,
                                                                 embedding, emb);
    }

    // ---- recurrence ----
    for (int t = 0; t < TM1; ++t) {
        attention_kernel<<<BB, 256, 0, stream>>>(h, enc, enc_proj, att_Wdec,
                                                 att_bdec, att_v, ctx);

        {   // pack gate-GEMM A: [emb_t | context | h]
            int slots = NMT * NKT_G * 32;
            pack_a_gates_f16<<<(slots + 127) / 128, 128, 0, stream>>>(emb, ctx, h, Ag, t);
        }
        wmma_gemm_f16<<<NNT_G, 32, 0, stream>>>(Ag, Bg, gbias,
                                                gates, (long long)GG, NKT_G);

        lstm_pointwise_kernel<<<(BB * HH + 255) / 256, 256, 0, stream>>>(gates, h, c);

        {   // pack fc-GEMM A from h_new
            int slots = NMT * NKT_FC * 32;
            pack_a_f16<<<(slots + 127) / 128, 128, 0, stream>>>(h, HH, Afc, NMT, NKT_FC);
        }
        // out[m, t, :] : base = out + t*V, row stride = (T-1)*V
        wmma_gemm_f16<<<NNT_FC, 32, 0, stream>>>(
            Afc, Bfc, fc_b, out + (size_t)t * VV, (long long)TM1 * VV, NKT_FC);
    }
}
